// RNN_12128987644612
// MI455X (gfx1250) — compile-verified, matrix operand order checked by reading, not run on hardware
//
#include <hip/hip_runtime.h>

// Bidirectional LSTM for MI455X (gfx1250, wave32).
// bf16 WMMA (v_wmma_f32_16x16x32_bf16) for both the input GEMM and the
// recurrent GEMM; f32 state/gates. Recurrence is weight-stationary:
// each wave holds its Wh tile (gate x K-half, 128 VGPRs) in registers for
// all 512 timesteps; per step only h (32KB, L2-resident) is re-loaded.
// Persistent grid with a device-scope atomic barrier per timestep and
// double-buffered h.

typedef __attribute__((ext_vector_type(16))) __bf16 v16bf;
typedef __attribute__((ext_vector_type(8)))  __bf16 v8bf;
typedef __attribute__((ext_vector_type(8)))  float  v8f;

#define B_  16
#define T_  512
#define I_  1024
#define U_  1024
#define N4  4096          // 4*U
#define M_  (B_*T_)       // 8192 rows of the input GEMM

// ---------------- workspace layout (bytes) ----------------
// [0,16M)    : x as bf16              [8192,1024]
// [16M,32M)  : WxT bf16 (fwd,bwd)     [4096,1024] each
// [32M,48M)  : WhT bf16 (fwd,bwd)     [4096,1024] each
// [48M,+128K): h double buffers       [2 dir][2 buf][16][1024] bf16
// [49M,+256B): barrier counters       [2 dir] (64B apart)
// [64M,320M) : xz f32 (fwd,bwd)       [8192,4096] each
#define XB_OFF    (0ull)
#define WXT_OFF   (16ull<<20)
#define WHT_OFF   (32ull<<20)
#define HBUF_OFF  (48ull<<20)
#define CTR_OFF   (49ull<<20)
#define XZ_OFF    (64ull<<20)

// Load a 16-bit A/B WMMA fragment from a row-major [*, stride-1024] bf16 matrix.
// Lane l<16 : row = l,    K = kt+0..7  and kt+16..23
// Lane l>=16: row = l-16, K = kt+8..15 and kt+24..31
__device__ __forceinline__ v16bf load_frag(const __bf16* p) {
    v8bf lo = *(const v8bf*)p;
    v8bf hi = *(const v8bf*)(p + 16);
    v16bf r;
#pragma unroll
    for (int i = 0; i < 8; ++i) { r[i] = lo[i]; r[i + 8] = hi[i]; }
    return r;
}

__device__ __forceinline__ float sigmoid_f(float x) {
    return 1.0f / (1.0f + __expf(-x));
}

// ---------------- pre-pass kernels ----------------
__global__ void f32_to_bf16_kernel(const float* __restrict__ in,
                                   __bf16* __restrict__ out, unsigned n) {
    unsigned i = blockIdx.x * blockDim.x + threadIdx.x;
    if (i < n) out[i] = (__bf16)in[i];
}

// W [1024 rows(k), 4096 cols(n)] f32  ->  WT [4096 rows(n), 1024 cols(k)] bf16
__global__ void transpose_bf16_kernel(const float* __restrict__ W,
                                      __bf16* __restrict__ WT) {
    __shared__ float tile[32][33];
    int c0 = blockIdx.x * 32;   // n
    int r0 = blockIdx.y * 32;   // k
    for (int i = threadIdx.y; i < 32; i += 8)
        tile[i][threadIdx.x] = W[(size_t)(r0 + i) * N4 + c0 + threadIdx.x];
    __syncthreads();
    for (int i = threadIdx.y; i < 32; i += 8)
        WT[(size_t)(c0 + i) * I_ + r0 + threadIdx.x] = (__bf16)tile[threadIdx.x][i];
}

// ---------------- phase 1: xz = x @ Wx + b (both directions) ----------------
// grid (N4/256, M/16, 2), block 128 (4 waves). Wave computes a 16x64 strip:
// one A fragment reused across 4 WMMA N-tiles per K step.
__global__ __launch_bounds__(128) void xz_gemm_kernel(
    const __bf16* __restrict__ xb,
    const __bf16* __restrict__ wxT_f, const __bf16* __restrict__ wxT_b,
    const float*  __restrict__ bias_f, const float* __restrict__ bias_b,
    float* __restrict__ xz_f, float* __restrict__ xz_b) {
    const int dir  = blockIdx.z;
    const __bf16* wT   = dir ? wxT_b : wxT_f;
    const float*  bias = dir ? bias_b : bias_f;
    float*        xz   = dir ? xz_b  : xz_f;

    const int wave = threadIdx.x >> 5;
    const int lane = threadIdx.x & 31;
    const int half = lane >> 4;
    const int l15  = lane & 15;
    const int mbase = blockIdx.y * 16;
    const int n0    = blockIdx.x * 256 + wave * 64;

    v8f acc[4];
#pragma unroll
    for (int j = 0; j < 4; ++j) acc[j] = (v8f){0,0,0,0,0,0,0,0};

    const __bf16* arow = xb + (size_t)(mbase + l15) * I_ + half * 8;
    for (int kt = 0; kt < I_; kt += 32) {
        v16bf a = load_frag(arow + kt);
#pragma unroll
        for (int j = 0; j < 4; ++j) {
            const __bf16* bp = wT + (size_t)(n0 + j * 16 + l15) * I_ + kt + half * 8;
            v16bf bfrg = load_frag(bp);
            acc[j] = __builtin_amdgcn_wmma_f32_16x16x32_bf16(
                false, a, false, bfrg, (short)0, acc[j], false, false);
        }
    }
#pragma unroll
    for (int j = 0; j < 4; ++j) {
        int col = n0 + j * 16 + l15;
        float bv = bias[col];
#pragma unroll
        for (int r = 0; r < 8; ++r) {
            int row = mbase + r + half * 8;
            xz[(size_t)row * N4 + col] = acc[j][r] + bv;
        }
    }
}

// ---------------- phase 2: recurrent scan (weight-stationary) ----------------
// grid (64, 2), block 256 (8 waves). Block owns 16 units [blk*16, +16).
// Wave wv = (gate g = wv>>1, K-half kh = wv&1): holds Wh tile
// [16 units x 512 K] in 16 v16bf register fragments for the whole scan.
// Per step: 16 chained WMMAs on the h fragment, partial z tiles combined
// through an 8KB LDS buffer, one (batch,unit) element of c/h state per thread
// kept in registers. One device-scope barrier per step, double-buffered h.
__global__ __launch_bounds__(256) void lstm_recur_kernel(
    const float*  __restrict__ xz_f, const float* __restrict__ xz_b,
    const __bf16* __restrict__ whT_f, const __bf16* __restrict__ whT_b,
    __bf16* __restrict__ h_buf,     // [2 dir][2 buf][16][1024]
    unsigned* __restrict__ ctr,     // [2 dir], 64B apart, zeroed each launch
    float* __restrict__ out) {
    __shared__ float zbuf[8][256];  // partial z tiles: [gate*2+khalf][m*16+n]

    const int dir  = blockIdx.y;
    const int blk  = blockIdx.x;        // 0..63
    const int tid  = threadIdx.x;
    const int wv   = tid >> 5;          // 0..7
    const int lane = tid & 31;
    const int half = lane >> 4;
    const int l15  = lane & 15;
    const int g    = wv >> 1;           // gate 0..3
    const int kh   = wv & 1;            // K half 0..1
    const int u0   = blk * 16;

    const float*  xz  = dir ? xz_b  : xz_f;
    const __bf16* whT = dir ? whT_b : whT_f;
    __bf16* hb0 = h_buf + (size_t)dir * (2 * B_ * U_);
    unsigned* cptr = ctr + dir * 16;

    // stationary B fragments: gate g rows (g*U + u0 + n), K in [kh*512, +512)
    v16bf breg[16];
    {
        const __bf16* brow = whT + (size_t)(g * U_ + u0 + l15) * U_ + kh * 512 + half * 8;
#pragma unroll
        for (int k16 = 0; k16 < 16; ++k16)
            breg[k16] = load_frag(brow + k16 * 32);
    }

    // this thread's (batch, unit) element of the 16x16 state tile
    const int em = tid >> 4;            // batch row 0..15
    const int en = tid & 15;            // unit col 0..15
    const int u  = u0 + en;
    float c_reg = 0.0f, h_reg = 0.0f;

    for (int step = 0; step < T_; ++step) {
        const int t = dir ? (T_ - 1 - step) : step;
        const __bf16* hread  = hb0 + (size_t)(step & 1) * (B_ * U_);
        __bf16*       hwrite = hb0 + (size_t)((step + 1) & 1) * (B_ * U_);

        // GEMM phase: z_partial[16x16] = h[16, kh*512..+512) @ Wh_tile
        v8f acc = (v8f){0,0,0,0,0,0,0,0};
        const __bf16* arow = hread + (size_t)l15 * U_ + kh * 512 + half * 8;
#pragma unroll
        for (int k16 = 0; k16 < 16; ++k16) {
            v16bf a = load_frag(arow + k16 * 32);
            acc = __builtin_amdgcn_wmma_f32_16x16x32_bf16(
                false, a, false, breg[k16], (short)0, acc, false, false);
        }
#pragma unroll
        for (int r = 0; r < 8; ++r)
            zbuf[wv][(r + 8 * half) * 16 + l15] = acc[r];
        __syncthreads();

        // elementwise phase: one (batch, unit) element per thread
        {
            size_t xrow = (size_t)(em * T_ + t) * N4;
            float zi = zbuf[0][tid] + zbuf[1][tid] + xz[xrow + 0 * U_ + u];
            float zf = zbuf[2][tid] + zbuf[3][tid] + xz[xrow + 1 * U_ + u];
            float zg = zbuf[4][tid] + zbuf[5][tid] + xz[xrow + 2 * U_ + u];
            float zo = zbuf[6][tid] + zbuf[7][tid] + xz[xrow + 3 * U_ + u];
            float ig = sigmoid_f(zi);
            float fg = sigmoid_f(zf);
            float gg = tanhf(zg);
            float og = sigmoid_f(zo);
            c_reg = fg * c_reg + ig * gg;
            h_reg = og * tanhf(c_reg);
            hwrite[(size_t)em * U_ + u] = (__bf16)h_reg;
            out[(size_t)(em * T_ + t) * (2 * U_) + dir * U_ + u] = h_reg;
        }

        // device-scope barrier across the 64 blocks of this direction
        __threadfence();
        __syncthreads();
        if (tid == 0) {
            __hip_atomic_fetch_add(cptr, 1u, __ATOMIC_RELEASE, __HIP_MEMORY_SCOPE_AGENT);
            unsigned target = 64u * (unsigned)(step + 1);
            while (__hip_atomic_load(cptr, __ATOMIC_ACQUIRE, __HIP_MEMORY_SCOPE_AGENT) < target) {
                __builtin_amdgcn_s_sleep(1);
            }
        }
        __syncthreads();
    }

    // final (h, c): out[SEQ .. SEQ+B*2U) = h, then c
    const size_t SEQ = (size_t)B_ * T_ * 2 * U_;
    out[SEQ + (size_t)em * (2 * U_) + dir * U_ + u] = h_reg;
    out[SEQ + (size_t)B_ * (2 * U_) + (size_t)em * (2 * U_) + dir * U_ + u] = c_reg;
}

// ---------------- host launch ----------------
extern "C" void kernel_launch(void* const* d_in, const int* in_sizes, int n_in,
                              void* d_out, int out_size, void* d_ws, size_t ws_size,
                              hipStream_t stream) {
    const float* x    = (const float*)d_in[0];
    const float* Wx_f = (const float*)d_in[1];
    const float* Wh_f = (const float*)d_in[2];
    const float* b_f  = (const float*)d_in[3];
    const float* Wx_b = (const float*)d_in[4];
    const float* Wh_b = (const float*)d_in[5];
    const float* b_b  = (const float*)d_in[6];
    float* out = (float*)d_out;

    char* ws = (char*)d_ws;
    __bf16*  xb     = (__bf16*)(ws + XB_OFF);
    __bf16*  wxT_f  = (__bf16*)(ws + WXT_OFF);
    __bf16*  wxT_b  = (__bf16*)(ws + WXT_OFF + (8ull << 20));
    __bf16*  whT_f  = (__bf16*)(ws + WHT_OFF);
    __bf16*  whT_b  = (__bf16*)(ws + WHT_OFF + (8ull << 20));
    __bf16*  h_buf  = (__bf16*)(ws + HBUF_OFF);
    unsigned* ctr   = (unsigned*)(ws + CTR_OFF);
    float*   xz_f   = (float*)(ws + XZ_OFF);
    float*   xz_b   = (float*)(ws + XZ_OFF + (128ull << 20));

    // zero h double-buffers (initial hidden state) and barrier counters
    hipMemsetAsync(h_buf, 0, 2ull * 2 * B_ * U_ * sizeof(__bf16), stream);
    hipMemsetAsync(ctr, 0, 256, stream);

    // x -> bf16
    {
        unsigned n = (unsigned)(B_ * T_ * I_);
        f32_to_bf16_kernel<<<(n + 255) / 256, 256, 0, stream>>>(x, xb, n);
    }
    // weights -> transposed bf16
    dim3 tg(N4 / 32, I_ / 32), tb(32, 8);
    transpose_bf16_kernel<<<tg, tb, 0, stream>>>(Wx_f, wxT_f);
    transpose_bf16_kernel<<<tg, tb, 0, stream>>>(Wx_b, wxT_b);
    transpose_bf16_kernel<<<tg, tb, 0, stream>>>(Wh_f, whT_f);
    transpose_bf16_kernel<<<tg, tb, 0, stream>>>(Wh_b, whT_b);

    // phase 1: input projections (both directions)
    xz_gemm_kernel<<<dim3(N4 / 256, M_ / 16, 2), 128, 0, stream>>>(
        xb, wxT_f, wxT_b, b_f, b_b, xz_f, xz_b);

    // phase 2: recurrence (persistent grid, per-step device barrier)
    lstm_recur_kernel<<<dim3(64, 2), 256, 0, stream>>>(
        xz_f, xz_b, whT_f, whT_b, h_buf, ctr, out);
}
